// QrotAttention_37409165149104
// MI455X (gfx1250) — compile-verified, matrix operand order checked by reading, not run on hardware
//
#include <hip/hip_runtime.h>

// ---------------------------------------------------------------------------
// Types matching the probe-verified gfx1250 WMMA builtin signatures
// ---------------------------------------------------------------------------
typedef __attribute__((ext_vector_type(16))) __bf16 v16bf;
typedef __attribute__((ext_vector_type(8)))  float  v8f;

// Exact parameter type of __builtin_amdgcn_global_load_async_to_lds_b128
// (from compiler diagnostic): non-const vector_size(16) int, AS1 / AS3.
typedef int v4i_vs __attribute__((__vector_size__(16)));
typedef __attribute__((address_space(1))) v4i_vs* gas_ptr;   // global
typedef __attribute__((address_space(3))) v4i_vs* las_ptr;   // LDS

#define C_DIM 1024
#define T_DIM 2048
#define B_DIM 4
#define M_DIM (B_DIM * T_DIM)   // 8192 token rows

// GEMM tiling: block tile 128(M) x 128(N), K-step 64.
// 256 threads = 8 waves arranged 4(Mrows) x 2(Ncols); each wave owns 32x64
// -> 2x4 accumulators = 16 WMMA per staged K-tile per wave.
#define BM 128
#define BN 128
#define KT 64
#define LDSK 72   // padded LDS row stride in bf16 (144B: 16B-aligned, 36-bank stride, conflict-free)

// CDNA5 async global->LDS path (ASYNCcnt); proven to exist on this toolchain.
#if __has_builtin(__builtin_amdgcn_global_load_async_to_lds_b128) && \
    __has_builtin(__builtin_amdgcn_s_wait_asynccnt)
#define HAVE_ASYNC_LDS 1
#else
#define HAVE_ASYNC_LDS 0
#endif

// ---------------------------------------------------------------------------
// f32 -> bf16 conversion
// ---------------------------------------------------------------------------
__global__ __launch_bounds__(256) void cvt_f32_bf16_kernel(const float* __restrict__ s,
                                                           __bf16* __restrict__ d, int n) {
  int i = blockIdx.x * 256 + threadIdx.x;
  if (i < n) d[i] = (__bf16)s[i];
}

// ---------------------------------------------------------------------------
// Fused: prev = (t>=offset ? q[t-offset] : identity); LN(concat(prev,q)) * g + b
// One 256-thread block per token row. Output bf16, row stride 2*C_DIM.
// ---------------------------------------------------------------------------
__global__ __launch_bounds__(256) void ln_concat_kernel(
    const float* __restrict__ q_old, int ldq,
    const float* __restrict__ identity,
    const float* __restrict__ gam, const float* __restrict__ beta,
    __bf16* __restrict__ hn, int offset) {
  __shared__ float rs[256];
  __shared__ float rq[256];
  const int row = blockIdx.x;
  const int t   = row & (T_DIM - 1);
  const int tid = threadIdx.x;

  const float* cur  = q_old + (size_t)row * ldq;
  const float* prev = (t >= offset) ? (q_old + (size_t)(row - offset) * ldq) : identity;

  const int e0 = tid * 8;                      // threads 0..127 -> prev, 128..255 -> cur
  const float* src = (e0 < C_DIM) ? (prev + e0) : (cur + (e0 - C_DIM));
  float4 v0 = ((const float4*)src)[0];
  float4 v1 = ((const float4*)src)[1];

  float s  = v0.x + v0.y + v0.z + v0.w + v1.x + v1.y + v1.z + v1.w;
  float sq = v0.x*v0.x + v0.y*v0.y + v0.z*v0.z + v0.w*v0.w
           + v1.x*v1.x + v1.y*v1.y + v1.z*v1.z + v1.w*v1.w;
  rs[tid] = s; rq[tid] = sq;
  __syncthreads();
  for (int o = 128; o > 0; o >>= 1) {
    if (tid < o) { rs[tid] += rs[tid + o]; rq[tid] += rq[tid + o]; }
    __syncthreads();
  }
  const float inv  = 1.0f / (2.0f * C_DIM);
  const float mean = rs[0] * inv;
  const float var  = rq[0] * inv - mean * mean;
  const float rstd = rsqrtf(var + 1e-5f);

  const float vals[8] = {v0.x, v0.y, v0.z, v0.w, v1.x, v1.y, v1.z, v1.w};
  __bf16* dst = hn + (size_t)row * (2 * C_DIM) + e0;
#pragma unroll
  for (int k = 0; k < 8; ++k) {
    float nv = (vals[k] - mean) * rstd * gam[e0 + k] + beta[e0 + k];
    dst[k] = (__bf16)nv;
  }
}

// ---------------------------------------------------------------------------
// Y = q * v (elementwise), output bf16 for the final WMMA projection.
// ---------------------------------------------------------------------------
__global__ __launch_bounds__(256) void mulqv_kernel(const float* __restrict__ q,
                                                    const float* __restrict__ v,
                                                    __bf16* __restrict__ y) {
  const size_t idx = (size_t)blockIdx.x * 256 + threadIdx.x;
  const size_t m = idx >> 10;
  const size_t c = idx & 1023;
  y[idx] = (__bf16)(q[(m << 10) + c] * v[(m << 11) + c]);
}

// ---------------------------------------------------------------------------
// bf16 WMMA GEMM:  out[M x N] = act( A[M x K] * W[N x K]^T + bias[N] )
// A, W row-major bf16; f32 accumulate via v_wmma_f32_16x16x32_bf16.
// Double-buffered LDS; async global->LDS (ASYNCcnt) when available, else
// register-prefetch + ds_store_b128. One workgroup barrier per K-tile.
// M mult of 128, N mult of 128, K mult of 64 (true for all calls here).
// ---------------------------------------------------------------------------
template <bool GELU_ACT, bool OUT_BF16>
__global__ __launch_bounds__(256) void gemm_bf16_wmma(
    const __bf16* __restrict__ A, int lda,
    const __bf16* __restrict__ W, int ldb,
    const float* __restrict__ bias,
    void* __restrict__ outp, int ldo, int K) {
  __shared__ __bf16 As[2][BM][LDSK];
  __shared__ __bf16 Bs[2][BN][LDSK];

  const int tid  = threadIdx.x;
  const int lane = tid & 31;
  const int wid  = tid >> 5;
  const int wr   = wid & 3;    // wave row group (32 M-rows)
  const int wc   = wid >> 2;   // wave col group (64 N-cols)
  const int blockM = blockIdx.y * BM;
  const int blockN = blockIdx.x * BN;

  const v8f vzero = {0.f, 0.f, 0.f, 0.f, 0.f, 0.f, 0.f, 0.f};
  v8f acc[2][4];
#pragma unroll
  for (int i = 0; i < 2; ++i)
#pragma unroll
    for (int j = 0; j < 4; ++j) acc[i][j] = vzero;

  // global->LDS staging assignment: each thread owns 32 contiguous elems of one row
  const int srow = tid >> 1;            // 0..127
  const int sseg = (tid & 1) * 32;      // 0 / 32
  const __bf16* gA = A + (size_t)(blockM + srow) * lda + sseg;
  const __bf16* gW = W + (size_t)(blockN + srow) * ldb + sseg;

  const int lg = lane >> 4;             // K-half selector per WMMA 16-bit layout
  const int lr = lane & 15;             // M/N index within fragment

  union Frag { v16bf v; uint2 u2[4]; };

  const int ntiles = K / KT;
  int buf = 0;

#if HAVE_ASYNC_LDS
  // ---- async prologue: tile 0 -> buffer 0 ------------------------------
#pragma unroll
  for (int u = 0; u < 4; ++u) {
    __builtin_amdgcn_global_load_async_to_lds_b128(
        (gas_ptr)(gA + u * 8), (las_ptr)&As[0][srow][sseg + u * 8], 0, 0);
    __builtin_amdgcn_global_load_async_to_lds_b128(
        (gas_ptr)(gW + u * 8), (las_ptr)&Bs[0][srow][sseg + u * 8], 0, 0);
  }
#else
  // ---- sync prologue: tile 0 -> registers ------------------------------
  uint4 ra[4], rb[4];
#pragma unroll
  for (int u = 0; u < 4; ++u) {
    ra[u] = *(const uint4*)(gA + u * 8);
    rb[u] = *(const uint4*)(gW + u * 8);
  }
#endif

  for (int kt = 0; kt < ntiles; ++kt) {
#if HAVE_ASYNC_LDS
    __builtin_amdgcn_s_wait_asynccnt(0);   // this wave's tile-kt LDS writes landed
    __syncthreads();                       // everyone's landed; buf^1 free to refill
    if (kt + 1 < ntiles) {
      const __bf16* pa = gA + (kt + 1) * KT;
      const __bf16* pw = gW + (kt + 1) * KT;
#pragma unroll
      for (int u = 0; u < 4; ++u) {
        __builtin_amdgcn_global_load_async_to_lds_b128(
            (gas_ptr)(pa + u * 8), (las_ptr)&As[buf ^ 1][srow][sseg + u * 8], 0, 0);
        __builtin_amdgcn_global_load_async_to_lds_b128(
            (gas_ptr)(pw + u * 8), (las_ptr)&Bs[buf ^ 1][srow][sseg + u * 8], 0, 0);
      }
    }
#else
    // commit current tile (in regs) to LDS, then prefetch next tile to regs
#pragma unroll
    for (int u = 0; u < 4; ++u) {
      *(uint4*)&As[buf][srow][sseg + u * 8] = ra[u];
      *(uint4*)&Bs[buf][srow][sseg + u * 8] = rb[u];
    }
    if (kt + 1 < ntiles) {
      const __bf16* pa = gA + (kt + 1) * KT;
      const __bf16* pw = gW + (kt + 1) * KT;
#pragma unroll
      for (int u = 0; u < 4; ++u) {
        ra[u] = *(const uint4*)(pa + u * 8);
        rb[u] = *(const uint4*)(pw + u * 8);
      }
    }
    __syncthreads();
#endif

    // ---- compute on buf: 2 K-substeps of 32, 8 WMMA each -----------------
#pragma unroll
    for (int ks = 0; ks < KT; ks += 32) {
      Frag a[2], b[4];
#pragma unroll
      for (int i = 0; i < 2; ++i) {
        const __bf16* r = &As[buf][wr * 32 + i * 16 + lr][ks];
        a[i].u2[0] = *(const uint2*)(r + lg * 8);
        a[i].u2[1] = *(const uint2*)(r + lg * 8 + 4);
        a[i].u2[2] = *(const uint2*)(r + 16 + lg * 8);
        a[i].u2[3] = *(const uint2*)(r + 16 + lg * 8 + 4);
      }
#pragma unroll
      for (int j = 0; j < 4; ++j) {
        const __bf16* r = &Bs[buf][wc * 64 + j * 16 + lr][ks];
        b[j].u2[0] = *(const uint2*)(r + lg * 8);
        b[j].u2[1] = *(const uint2*)(r + lg * 8 + 4);
        b[j].u2[2] = *(const uint2*)(r + 16 + lg * 8);
        b[j].u2[3] = *(const uint2*)(r + 16 + lg * 8 + 4);
      }
#pragma unroll
      for (int i = 0; i < 2; ++i)
#pragma unroll
        for (int j = 0; j < 4; ++j)
          acc[i][j] = __builtin_amdgcn_wmma_f32_16x16x32_bf16(
              false, a[i].v, false, b[j].v, (short)0, acc[i][j], false, false);
    }

#if !HAVE_ASYNC_LDS
    __syncthreads();   // all reads of buf done before it is overwritten (kt+2)
#endif
    buf ^= 1;
  }

  // Epilogue. C/D layout: VGPR p -> (M = p + 8*(lane>>4), N = lane&15).
#pragma unroll
  for (int i = 0; i < 2; ++i) {
#pragma unroll
    for (int j = 0; j < 4; ++j) {
      const int col = blockN + wc * 64 + j * 16 + lr;
      const float bs = bias[col];
#pragma unroll
      for (int p = 0; p < 8; ++p) {
        const int row = blockM + wr * 32 + i * 16 + p + 8 * lg;
        float v = acc[i][j][p] + bs;
        if (GELU_ACT) v = 0.5f * v * (1.0f + erff(v * 0.70710678118654752f));
        if (OUT_BF16)
          ((__bf16*)outp)[(size_t)row * ldo + col] = (__bf16)v;
        else
          ((float*)outp)[(size_t)row * ldo + col] = v;
      }
    }
  }
}

// ---------------------------------------------------------------------------
// Host orchestration
// ---------------------------------------------------------------------------
extern "C" void kernel_launch(void* const* d_in, const int* in_sizes, int n_in,
                              void* d_out, int out_size, void* d_ws, size_t ws_size,
                              hipStream_t stream) {
  const float* x        = (const float*)d_in[0];
  const float* W_attn2  = (const float*)d_in[1];
  const float* b_attn2  = (const float*)d_in[2];
  const float* W_proj   = (const float*)d_in[3];
  const float* b_proj   = (const float*)d_in[4];
  const float* ln_g     = (const float*)d_in[5];
  const float* ln_b     = (const float*)d_in[6];
  const float* W1       = (const float*)d_in[7];
  const float* b1       = (const float*)d_in[8];
  const float* W2       = (const float*)d_in[9];
  const float* b2       = (const float*)d_in[10];
  const float* identity = (const float*)d_in[11];

  const int M = M_DIM, C = C_DIM;

  // Workspace carving (~287 MB total, fixed sizes).
  char* p = (char*)d_ws;
  auto carve = [&](size_t bytes) -> void* {
    void* r = (void*)p;
    p += (bytes + 255) & ~(size_t)255;
    return r;
  };
  __bf16* xb  = (__bf16*)carve((size_t)M * C * 2);            // 16 MB
  __bf16* Wab = (__bf16*)carve((size_t)2 * C * C * 2);        //  4 MB
  __bf16* W1b = (__bf16*)carve((size_t)4 * C * 2 * C * 2);    // 16 MB
  __bf16* W2b = (__bf16*)carve((size_t)C * 4 * C * 2);        //  8 MB
  __bf16* Wpb = (__bf16*)carve((size_t)C * C * 2);            //  2 MB
  float*  qv  = (float*)carve((size_t)M * 2 * C * 4);         // 64 MB (q | v)
  float*  qA  = (float*)carve((size_t)M * C * 4);             // 32 MB
  float*  qB  = (float*)carve((size_t)M * C * 4);             // 32 MB
  __bf16* hn  = (__bf16*)carve((size_t)M * 2 * C * 2);        // 32 MB
  __bf16* hb  = (__bf16*)carve((size_t)M * 4 * C * 2);        // 64 MB
  __bf16* yb  = (__bf16*)carve((size_t)M * C * 2);            // 16 MB
  if ((size_t)(p - (char*)d_ws) > ws_size) return;            // workspace too small

  // 1) bf16 conversions (weights + x)
  auto cvt = [&](const float* s, __bf16* d, int n) {
    cvt_f32_bf16_kernel<<<dim3((n + 255) / 256), dim3(256), 0, stream>>>(s, d, n);
  };
  cvt(x,       xb,  M * C);
  cvt(W_attn2, Wab, 2 * C * C);
  cvt(W1,      W1b, 4 * C * 2 * C);
  cvt(W2,      W2b, C * 4 * C);
  cvt(W_proj,  Wpb, C * C);

  // 2) qv = x @ W_attn2^T + b_attn2 -> f32, row stride 2C (q = qv[:, :C], v = qv[:, C:])
  gemm_bf16_wmma<false, false><<<dim3(2 * C / BN, M / BM), dim3(256), 0, stream>>>(
      xb, C, Wab, C, b_attn2, qv, 2 * C, C);

  // 3) combine loop: 11 doublings
  const float* q_old = qv;
  int ld_old = 2 * C;
  float* qbufs[2] = {qA, qB};
  int cur = 0;
  for (int offset = 1; offset < T_DIM; offset <<= 1) {
    ln_concat_kernel<<<dim3(M), dim3(256), 0, stream>>>(q_old, ld_old, identity,
                                                        ln_g, ln_b, hn, offset);
    // h = gelu(hn @ W1^T + b1)  -> bf16 [M x 4C]
    gemm_bf16_wmma<true, true><<<dim3(4 * C / BN, M / BM), dim3(256), 0, stream>>>(
        hn, 2 * C, W1b, 2 * C, b1, hb, 4 * C, 2 * C);
    // q_new = h @ W2^T + b2     -> f32 [M x C]
    gemm_bf16_wmma<false, false><<<dim3(C / BN, M / BM), dim3(256), 0, stream>>>(
        hb, 4 * C, W2b, 4 * C, b2, qbufs[cur], C, 4 * C);
    q_old = qbufs[cur];
    ld_old = C;
    cur ^= 1;
  }

  // 4) Y = q * v (bf16), then out = Y @ W_proj^T + b_proj (f32 into d_out)
  mulqv_kernel<<<dim3(M * C / 256), dim3(256), 0, stream>>>(q_old, qv + C, yb);
  gemm_bf16_wmma<false, false><<<dim3(C / BN, M / BM), dim3(256), 0, stream>>>(
      yb, C, Wpb, C, b_proj, d_out, C, C);
}
